// CustomLSTM_74715251081424
// MI455X (gfx1250) — matrix operand order repaired and checked by reading, not compiled, not run on hardware
//
#include <hip/hip_runtime.h>
#include <hip/hip_bf16.h>
#include <math.h>

// ---------------------------------------------------------------------------
// Highway-LSTM for MI455X (gfx1250, wave32, WMMA).
//  B=32, T=2048, I=256, H=256.  Phase 1: bf16 WMMA GEMM pi = x*Wi^T + bi.
//  Phase 2: persistent 16-WG recurrence, bf16 WMMA h*Ws^T with fp32 state.
// ---------------------------------------------------------------------------

typedef __bf16 bf16_t;
typedef __attribute__((ext_vector_type(16))) __bf16 v16bf;
typedef __attribute__((ext_vector_type(8)))  __bf16 v8bf;
typedef __attribute__((ext_vector_type(8)))  float  v8f;
typedef __attribute__((ext_vector_type(4)))  float  v4f;

#define LSTM_B 32
#define LSTM_T 2048
#define LSTM_I 256
#define LSTM_H 256
#define G6     (6 * LSTM_H)   // 1536
#define NWG_REC 16            // recurrence workgroups (16 * 16 cols = 256 = H)

static __device__ __forceinline__ v16bf cat16(v8bf lo, v8bf hi) {
  return __builtin_shufflevector(lo, hi, 0, 1, 2, 3, 4, 5, 6, 7,
                                         8, 9, 10, 11, 12, 13, 14, 15);
}

static __device__ __forceinline__ float sigmoidf_(float x) {
  return 1.0f / (1.0f + __expf(-x));
}
static __device__ __forceinline__ float tanhf_(float x) {
  x = fminf(fmaxf(x, -15.0f), 15.0f);
  float e = __expf(2.0f * x);
  return (e - 1.0f) / (e + 1.0f);
}

// ---------------------------------------------------------------------------
// Kernel 0: zero output region [0, B*T*H), zero h double-buffer, reset barrier.
// ---------------------------------------------------------------------------
__global__ void lstm_init_kernel(float* __restrict__ out, size_t nout,
                                 bf16_t* __restrict__ hbuf, int* __restrict__ cnt) {
  size_t i = (size_t)blockIdx.x * blockDim.x + threadIdx.x;
  size_t stride = (size_t)gridDim.x * blockDim.x;
  for (size_t k = i; k < nout; k += stride) out[k] = 0.0f;
  for (size_t k = i; k < (size_t)2 * LSTM_B * LSTM_H; k += stride)
    hbuf[k] = (bf16_t)0.0f;
  if (i == 0) *cnt = 0;
}

// ---------------------------------------------------------------------------
// Kernel 1: pi[t][b][g] = sum_i x[b][t][i] * Wi[g][i] + bi[g]
// Grid: (T, 1536/128). Block: 256 threads = 8 waves.
// Macro tile: M=32 (all b for fixed t), N=128, K chunked by 32.
// Wave w: mtile = w&1 (16 rows), npair = w>>1 -> two 16-wide N tiles.
// ---------------------------------------------------------------------------
__global__ __launch_bounds__(256) void lstm_proj_kernel(
    const float* __restrict__ x, const float* __restrict__ Wi,
    const float* __restrict__ bi, float* __restrict__ pi) {
  const int t  = blockIdx.x;
  const int n0 = blockIdx.y * 128;
  const int tid = threadIdx.x;
  const int lane = tid & 31;
  const int wave = tid >> 5;
  const int mt = wave & 1;
  const int np = wave >> 1;

  __shared__ __align__(16) bf16_t As[32][40];   // [M][Kchunk+pad]
  __shared__ __align__(16) bf16_t Bs[128][40];  // [N][Kchunk+pad] (Wi rows)

  v8f acc0 = {0.f, 0.f, 0.f, 0.f, 0.f, 0.f, 0.f, 0.f};
  v8f acc1 = {0.f, 0.f, 0.f, 0.f, 0.f, 0.f, 0.f, 0.f};

  const int n_lane = lane & 15;
  const int hi_half = (lane >= 16);

  for (int kb = 0; kb < LSTM_I; kb += 32) {
    __syncthreads();
    // ---- stage A: x[b][t][kb..kb+31] -> bf16, 1 float4 per thread
    {
      int row = tid >> 3;
      int c4  = (tid & 7) * 4;
      const float* src = x + ((size_t)row * LSTM_T + t) * LSTM_I + kb + c4;
      v4f l = *(const v4f*)src;
      As[row][c4 + 0] = (bf16_t)l[0];
      As[row][c4 + 1] = (bf16_t)l[1];
      As[row][c4 + 2] = (bf16_t)l[2];
      As[row][c4 + 3] = (bf16_t)l[3];
    }
    // ---- stage B: Wi[n0+row][kb..kb+31] -> bf16, 4 float4 per thread
#pragma unroll
    for (int e = 0; e < 4; ++e) {
      int lin = tid + e * 256;       // float4 index among 128*8
      int row = lin >> 3;
      int c4  = (lin & 7) * 4;
      const float* src = Wi + (size_t)(n0 + row) * LSTM_I + kb + c4;
      v4f l = *(const v4f*)src;
      Bs[row][c4 + 0] = (bf16_t)l[0];
      Bs[row][c4 + 1] = (bf16_t)l[1];
      Bs[row][c4 + 2] = (bf16_t)l[2];
      Bs[row][c4 + 3] = (bf16_t)l[3];
    }
    __syncthreads();

    // A fragment (16x32 bf16): lanes<16 hold K0-15 of row lane;
    //                          lanes>=16 hold K16-31.
    int ar = mt * 16 + n_lane;
    int ak = hi_half ? 16 : 0;
    v16bf a = cat16(*(const v8bf*)&As[ar][ak],
                    *(const v8bf*)&As[ar][ak + 8]);
    // B fragments (32x16): lane holds column n, contiguous 16 K values.
    int koff = hi_half ? 16 : 0;
    int br0 = np * 32 + n_lane;
    v16bf b0 = cat16(*(const v8bf*)&Bs[br0][koff],
                     *(const v8bf*)&Bs[br0][koff + 8]);
    v16bf b1 = cat16(*(const v8bf*)&Bs[br0 + 16][koff],
                     *(const v8bf*)&Bs[br0 + 16][koff + 8]);

    acc0 = __builtin_amdgcn_wmma_f32_16x16x32_bf16(false, a, false, b0,
                                                   (short)0, acc0, false, false);
    acc1 = __builtin_amdgcn_wmma_f32_16x16x32_bf16(false, a, false, b1,
                                                   (short)0, acc1, false, false);
  }

  // epilogue: C layout -> acc[r] is (M = r + 8*hi_half, N = lane&15)
  const int rowadd = hi_half ? 8 : 0;
#pragma unroll
  for (int half = 0; half < 2; ++half) {
    v8f acc = half ? acc1 : acc0;
    int g = n0 + np * 32 + half * 16 + n_lane;
    float bias = bi[g];
#pragma unroll
    for (int r = 0; r < 8; ++r) {
      int b = mt * 16 + r + rowadd;
      pi[((size_t)t * LSTM_B + b) * G6 + g] = acc[r] + bias;
    }
  }
}

// ---------------------------------------------------------------------------
// Kernel 2: persistent recurrence. 16 WGs x 64 threads (2 waves).
// WG wg owns h-columns [wg*16, wg*16+16) across all 5 gates; wave = M tile.
// Each lane holds 8 (b,j) cells: c/h state in fp32 registers for all T steps.
// h exchanged via double-buffered bf16 global buffer + 1 grid barrier/step.
// ---------------------------------------------------------------------------
__global__ __launch_bounds__(64) void lstm_rec_kernel(
    const float* __restrict__ pi, const int* __restrict__ lengths,
    const float* __restrict__ Ws, const float* __restrict__ bs,
    float* __restrict__ out, bf16_t* __restrict__ hbuf, int* __restrict__ cnt) {
  const int jbase = blockIdx.x * 16;
  const int tid = threadIdx.x;
  const int lane = tid & 31;
  const int mt = tid >> 5;                 // wave id = M tile (0/1)
  const int n_lane = lane & 15;
  const int hi_half = (lane >= 16);
  const int rowadd = hi_half ? 8 : 0;
  const int jglob = jbase + n_lane;

  // Ws slice for this WG: rows g*256 + jbase + n (g=0..4, n=0..15) as bf16
  __shared__ __align__(16) bf16_t Wlds[80][264];  // [gate*16+n][K + pad]
  for (int idx = tid; idx < 80 * 64; idx += 64) { // 64 float4 chunks per row
    int rr = idx >> 6;
    int c4 = (idx & 63) * 4;
    int g = rr >> 4, n = rr & 15;
    const float* src = Ws + ((size_t)(g * LSTM_H + jbase + n)) * LSTM_H + c4;
    v4f l = *(const v4f*)src;
    Wlds[rr][c4 + 0] = (bf16_t)l[0];
    Wlds[rr][c4 + 1] = (bf16_t)l[1];
    Wlds[rr][c4 + 2] = (bf16_t)l[2];
    Wlds[rr][c4 + 3] = (bf16_t)l[3];
  }

  float bsv[5];
#pragma unroll
  for (int g = 0; g < 5; ++g) bsv[g] = bs[g * LSTM_H + jglob];

  int   lenv[8];
  float hreg[8], creg[8];
#pragma unroll
  for (int r = 0; r < 8; ++r) {
    int b = mt * 16 + r + rowadd;
    lenv[r] = lengths[b];
    hreg[r] = 0.0f;
    creg[r] = 0.0f;
  }
  const int Tmax = lengths[0];  // sorted descending -> max length
  __syncthreads();

  int phase = 0;  // used by tid==0 only
  for (int t = 0; t < Tmax; ++t) {
    const bf16_t* hcur = hbuf + (size_t)(t & 1) * (LSTM_B * LSTM_H);
    bf16_t*       hnxt = hbuf + (size_t)((t + 1) & 1) * (LSTM_B * LSTM_H);

    v8f acc[5];
    const v8f vz = {0.f, 0.f, 0.f, 0.f, 0.f, 0.f, 0.f, 0.f};
#pragma unroll
    for (int g = 0; g < 5; ++g) acc[g] = vz;

    if (t + 1 < Tmax)  // pull next step's pi rows toward L2 (global_prefetch_b8)
      __builtin_prefetch(pi + (size_t)(t + 1) * (LSTM_B * G6) +
                             (size_t)(mt * 16 + rowadd) * G6 + jglob, 0, 1);

#pragma unroll
    for (int kb = 0; kb < LSTM_H; kb += 32) {
      // A fragment straight from global bf16 h buffer (L2-resident, 16B loads)
      int ar = mt * 16 + n_lane;
      int ak = kb + (hi_half ? 16 : 0);
      v16bf a = cat16(*(const v8bf*)(hcur + (size_t)ar * LSTM_H + ak),
                      *(const v8bf*)(hcur + (size_t)ar * LSTM_H + ak + 8));
      int koff = kb + (hi_half ? 16 : 0);
#pragma unroll
      for (int g = 0; g < 5; ++g) {
        v16bf bf = cat16(*(const v8bf*)&Wlds[g * 16 + n_lane][koff],
                         *(const v8bf*)&Wlds[g * 16 + n_lane][koff + 8]);
        acc[g] = __builtin_amdgcn_wmma_f32_16x16x32_bf16(
            false, a, false, bf, (short)0, acc[g], false, false);
      }
    }

    // Elementwise gate math fully in registers (identical lane mapping across
    // all 5 accumulators): acc[g][r] is (b = mt*16+r+rowadd, j = jglob).
    const float* pit = pi + (size_t)t * (LSTM_B * G6);
#pragma unroll
    for (int r = 0; r < 8; ++r) {
      int b = mt * 16 + r + rowadd;
      const float* pb = pit + (size_t)b * G6 + jglob;
      float gi = sigmoidf_(acc[0][r] + bsv[0] + pb[0 * LSTM_H]);
      float gf = sigmoidf_(acc[1][r] + bsv[1] + pb[1 * LSTM_H]);
      float gm = tanhf_   (acc[2][r] + bsv[2] + pb[2 * LSTM_H]);
      float go = sigmoidf_(acc[3][r] + bsv[3] + pb[3 * LSTM_H]);
      float hw = sigmoidf_(acc[4][r] + bsv[4] + pb[4 * LSTM_H]);
      float byp = pb[5 * LSTM_H];

      float cn = gi * gm + gf * creg[r];
      float ov = go * tanhf_(cn);
      ov = hw * ov + (1.0f - hw) * byp;

      bool active = (t < lenv[r]);
      out[((size_t)b * LSTM_T + t) * LSTM_H + jglob] = active ? ov : 0.0f;
      if (active) { creg[r] = cn; hreg[r] = ov; }
      // always publish current h (carries old state for inactive rows)
      hnxt[(size_t)b * LSTM_H + jglob] = (bf16_t)hreg[r];
    }

    // ---- grid-wide barrier (single per step; double buffer removes races)
    __syncthreads();
    if (tid == 0) {
      __threadfence();
      __hip_atomic_fetch_add(cnt, 1, __ATOMIC_RELEASE, __HIP_MEMORY_SCOPE_AGENT);
      ++phase;
      const int target = phase * NWG_REC;
      while (__hip_atomic_load(cnt, __ATOMIC_ACQUIRE,
                               __HIP_MEMORY_SCOPE_AGENT) < target) {
        __builtin_amdgcn_s_sleep(1);
      }
    }
    __syncthreads();
  }

  // Final states from exact fp32 registers.
  const size_t outY = (size_t)LSTM_B * LSTM_T * LSTM_H;
#pragma unroll
  for (int r = 0; r < 8; ++r) {
    int b = mt * 16 + r + rowadd;
    out[outY + (size_t)b * LSTM_H + jglob] = hreg[r];
    out[outY + (size_t)LSTM_B * LSTM_H + (size_t)b * LSTM_H + jglob] = creg[r];
  }
}

// ---------------------------------------------------------------------------
// Launch
// ---------------------------------------------------------------------------
extern "C" void kernel_launch(void* const* d_in, const int* in_sizes, int n_in,
                              void* d_out, int out_size, void* d_ws, size_t ws_size,
                              hipStream_t stream) {
  (void)in_sizes; (void)n_in; (void)out_size; (void)ws_size;
  const float* x       = (const float*)d_in[0];
  const int*   lengths = (const int*)d_in[1];
  const float* Wi      = (const float*)d_in[2];
  const float* bi      = (const float*)d_in[3];
  const float* Ws      = (const float*)d_in[4];
  const float* bs      = (const float*)d_in[5];
  float* out = (float*)d_out;

  // Workspace layout:
  //   [0, 4)          : grid-barrier counter (reset every launch)
  //   [1024, +32KB)   : double-buffered bf16 h state (2 x 32x256)
  //   [64KB, +384MB)  : pi_all fp32 [T][B][6H]
  char* ws = (char*)d_ws;
  int*    cnt  = (int*)ws;
  bf16_t* hbuf = (bf16_t*)(ws + 1024);
  float*  pi   = (float*)(ws + 65536);

  const size_t nout_y = (size_t)LSTM_B * LSTM_T * LSTM_H;
  lstm_init_kernel<<<4096, 256, 0, stream>>>(out, nout_y, hbuf, cnt);

  dim3 pgrid(LSTM_T, G6 / 128);
  lstm_proj_kernel<<<pgrid, 256, 0, stream>>>(x, Wi, bi, pi);

  lstm_rec_kernel<<<NWG_REC, 64, 0, stream>>>(pi, lengths, Ws, bs, out, hbuf, cnt);
}
